// GCN_11665131176089
// MI455X (gfx1250) — matrix operand order repaired and checked by reading, not compiled, not run on hardware
//
#include <hip/hip_runtime.h>
#include <math.h>

#define BB 8
#define NN 2048
#define DD 256

typedef __bf16 bf16_t;
typedef __attribute__((ext_vector_type(8)))  float  f32x8;
typedef __attribute__((ext_vector_type(8)))  bf16_t bf16x8;
typedef __attribute__((ext_vector_type(16))) bf16_t bf16x16;
typedef __attribute__((ext_vector_type(8)))  float  v8f;

// assemble a 16-element bf16 WMMA fragment from two 8-element (b128) loads
__device__ __forceinline__ bf16x16 mkfrag(const bf16_t* lo, const bf16_t* hi) {
  bf16x8 l = *(const bf16x8*)lo;
  bf16x8 h = *(const bf16x8*)hi;
  return __builtin_shufflevector(l, h, 0,1,2,3,4,5,6,7,8,9,10,11,12,13,14,15);
}

// ---- wave-level GEMM: one 32x64 output strip, K-loop in steps of 32 --------
// A: row-major [M][K] bf16, Bt: row-major [Ncols][K] bf16 (i.e. B^T)
// 2 A-fragments x 4 B-fragments -> 8 WMMAs per K-step (8 accumulators)
__device__ __forceinline__ void wmma_tile8(const bf16_t* __restrict__ A,
                                           const bf16_t* __restrict__ Bt,
                                           int lda, int ldb, int K,
                                           int row0, int col0, int lane,
                                           v8f acc[8]) {
  const int q = lane & 15, h = lane >> 4;
  // A fragment: lanes 0-15 hold K = {0..7, 16..23}; lanes 16-31 K = {8..15, 24..31}
  const bf16_t* pa0 = A + (size_t)(row0 + q) * lda + h * 8;
  const bf16_t* pa1 = pa0 + (size_t)16 * lda;
  // B fragment: lanes 0-15 hold K = {0..7, 16..23}; lanes 16-31 K = {8..15, 24..31}
  const bf16_t* pb = Bt + (size_t)(col0 + q) * ldb + h * 8;
#pragma unroll 2
  for (int k0 = 0; k0 < K; k0 += 32) {
    bf16x16 a0 = mkfrag(pa0 + k0, pa0 + k0 + 16);
    bf16x16 a1 = mkfrag(pa1 + k0, pa1 + k0 + 16);
#pragma unroll
    for (int j = 0; j < 4; ++j) {
      const bf16_t* pbj = pb + (size_t)j * 16 * ldb + k0;
      bf16x16 bfrag = mkfrag(pbj, pbj + 16);
      acc[j]     = __builtin_amdgcn_wmma_f32_16x16x32_bf16(
          false, a0, false, bfrag, (short)0, acc[j],     false, false);
      acc[4 + j] = __builtin_amdgcn_wmma_f32_16x16x32_bf16(
          false, a1, false, bfrag, (short)0, acc[4 + j], false, false);
    }
  }
}

// ---- d_inv_sqrt = rsqrt(rowsum(adj) + 1e-30), one wave per row -------------
__global__ void k_rowsum(const float* __restrict__ adj, float* __restrict__ dinv) {
  int wave = blockIdx.x * 8 + (threadIdx.x >> 5);
  int lane = threadIdx.x & 31;
  const float* row = adj + (size_t)wave * NN;
  float s = 0.f;
  for (int i = lane; i < NN; i += 32) s += row[i];
#pragma unroll
  for (int m = 16; m > 0; m >>= 1) s += __shfl_xor(s, m, 32);
  if (lane == 0) dinv[wave] = rsqrtf(s + 1e-30f);
}

__global__ void k_copy4(const float4* __restrict__ src, float4* __restrict__ dst, int n4) {
  int i = blockIdx.x * blockDim.x + threadIdx.x;
  if (i < n4) dst[i] = src[i];
}

// adj_bf = bf16(adj), 8 elements per thread (2x b128 load, 1x b128 store)
__global__ void k_pack_adj(const float* __restrict__ adj, bf16_t* __restrict__ adj_bf) {
  size_t i = ((size_t)blockIdx.x * blockDim.x + threadIdx.x) * 8;
  f32x8 v = *(const f32x8*)(adj + i);
  bf16x8 r;
#pragma unroll
  for (int k = 0; k < 8; ++k) r[k] = (bf16_t)v[k];
  *(bf16x8*)(adj_bf + i) = r;
}

// wt[e][d] = bf16(W[d][e])   (B^T layout for GEMM2)
__global__ void k_pack_w(const float* __restrict__ W, bf16_t* __restrict__ wt) {
  int i = blockIdx.x * blockDim.x + threadIdx.x;
  int e = i >> 8, d = i & 255;
  wt[i] = (bf16_t)W[d * DD + e];
}

// xs_t[b][d][m] = bf16(dinv[b][m] * X[b][m][d])   (pre-scaled, transposed feats)
__global__ void k_pack_xs(const float* __restrict__ X, const float* __restrict__ dinv,
                          bf16_t* __restrict__ xs_t) {
  int i = blockIdx.x * blockDim.x + threadIdx.x;
  int b = i / (NN * DD);
  int rem = i - b * NN * DD;
  int m = rem / DD, d = rem - m * DD;
  xs_t[(size_t)b * DD * NN + (size_t)d * NN + m] =
      (bf16_t)(dinv[b * NN + m] * X[i]);
}

// tmp[b][n][d] = dinv[n] * sum_m adj[n][m] * xs[m][d]; store row- and col-major
__global__ void k_gemm_lap(const bf16_t* __restrict__ adj_bf, const bf16_t* __restrict__ xs_t,
                           const float* __restrict__ dinv,
                           bf16_t* __restrict__ tmp, bf16_t* __restrict__ tmpT) {
  int wave = blockIdx.x * 8 + (threadIdx.x >> 5);
  int lane = threadIdx.x & 31;
  const int tiles_n = DD / 64, per_batch = (NN / 32) * tiles_n;
  int b = wave / per_batch, r = wave % per_batch;
  int row0 = (r / tiles_n) * 32, col0 = (r % tiles_n) * 64;
  v8f acc[8] = {};
  wmma_tile8(adj_bf + (size_t)b * NN * NN, xs_t + (size_t)b * DD * NN,
             NN, NN, NN, row0, col0, lane, acc);
  int c = lane & 15, h = lane >> 4;
#pragma unroll
  for (int a = 0; a < 2; ++a)
#pragma unroll
    for (int j = 0; j < 4; ++j)
#pragma unroll
      for (int rr = 0; rr < 8; ++rr) {
        int row = row0 + a * 16 + h * 8 + rr, col = col0 + j * 16 + c;
        float v = acc[a * 4 + j][rr] * dinv[b * NN + row];
        tmp [(size_t)b * NN * DD + (size_t)row * DD + col] = (bf16_t)v;
        tmpT[(size_t)b * DD * NN + (size_t)col * NN + row] = (bf16_t)v;
      }
}

// h[b][n][e] = sum_d tmp[n][d] * W[d][e]
__global__ void k_gemm_w(const bf16_t* __restrict__ tmp, const bf16_t* __restrict__ wt,
                         bf16_t* __restrict__ hbuf) {
  int wave = blockIdx.x * 8 + (threadIdx.x >> 5);
  int lane = threadIdx.x & 31;
  const int tiles_n = DD / 64, per_batch = (NN / 32) * tiles_n;
  int b = wave / per_batch, r = wave % per_batch;
  int row0 = (r / tiles_n) * 32, col0 = (r % tiles_n) * 64;
  v8f acc[8] = {};
  wmma_tile8(tmp + (size_t)b * NN * DD, wt, DD, DD, DD, row0, col0, lane, acc);
  int c = lane & 15, h = lane >> 4;
#pragma unroll
  for (int a = 0; a < 2; ++a)
#pragma unroll
    for (int j = 0; j < 4; ++j)
#pragma unroll
      for (int rr = 0; rr < 8; ++rr) {
        int row = row0 + a * 16 + h * 8 + rr, col = col0 + j * 16 + c;
        hbuf[(size_t)b * NN * DD + (size_t)row * DD + col] = (bf16_t)acc[a * 4 + j][rr];
      }
}

// scores[b][n][m] = leaky_relu(h[n]·h[m], 0.2) * adj[n][m]
__global__ void k_gemm_scores(const bf16_t* __restrict__ hbuf, const float* __restrict__ adj,
                              float* __restrict__ scores) {
  int wave = blockIdx.x * 8 + (threadIdx.x >> 5);
  int lane = threadIdx.x & 31;
  const int tiles_n = NN / 64, per_batch = (NN / 32) * tiles_n;
  int b = wave / per_batch, r = wave % per_batch;
  int row0 = (r / tiles_n) * 32, col0 = (r % tiles_n) * 64;
  const bf16_t* hb = hbuf + (size_t)b * NN * DD;
  v8f acc[8] = {};
  wmma_tile8(hb, hb, DD, DD, DD, row0, col0, lane, acc);  // B^T == h itself
  int c = lane & 15, h = lane >> 4;
#pragma unroll
  for (int a = 0; a < 2; ++a)
#pragma unroll
    for (int j = 0; j < 4; ++j)
#pragma unroll
      for (int rr = 0; rr < 8; ++rr) {
        int row = row0 + a * 16 + h * 8 + rr, col = col0 + j * 16 + c;
        float v = acc[a * 4 + j][rr];
        v = (v > 0.f) ? v : 0.2f * v;
        size_t idx = (size_t)b * NN * NN + (size_t)row * NN + col;
        scores[idx] = v * adj[idx];
      }
}

// row softmax; writes bf16 attn in-place over the f32 row (pitch 2*NN bf16)
__global__ void k_softmax(float* __restrict__ scores) {
  __shared__ float red[256];
  float* srow = scores + (size_t)blockIdx.x * NN;
  bf16_t* arow = (bf16_t*)srow;
  int t = threadIdx.x;
  float v[8], mx = -1e30f;
#pragma unroll
  for (int j = 0; j < 8; ++j) { v[j] = srow[t + 256 * j]; mx = fmaxf(mx, v[j]); }
  red[t] = mx; __syncthreads();
  for (int s = 128; s > 0; s >>= 1) { if (t < s) red[t] = fmaxf(red[t], red[t + s]); __syncthreads(); }
  mx = red[0]; __syncthreads();
  float sum = 0.f;
#pragma unroll
  for (int j = 0; j < 8; ++j) { v[j] = __expf(v[j] - mx); sum += v[j]; }
  red[t] = sum; __syncthreads();
  for (int s = 128; s > 0; s >>= 1) { if (t < s) red[t] += red[t + s]; __syncthreads(); }
  float inv = 1.0f / red[0];
  __syncthreads();
#pragma unroll
  for (int j = 0; j < 8; ++j) arow[t + 256 * j] = (bf16_t)(v[j] * inv);
}

// out[b][n][d] = tanh(sum_m attn[n][m]*tmp[m][d] + bias[d])
__global__ void k_gemm_attn(const float* __restrict__ scores, const bf16_t* __restrict__ tmpT,
                            const float* __restrict__ bias, float* __restrict__ out) {
  int wave = blockIdx.x * 8 + (threadIdx.x >> 5);
  int lane = threadIdx.x & 31;
  const int tiles_n = DD / 64, per_batch = (NN / 32) * tiles_n;
  int b = wave / per_batch, r = wave % per_batch;
  int row0 = (r / tiles_n) * 32, col0 = (r % tiles_n) * 64;
  const bf16_t* attn = (const bf16_t*)(scores + (size_t)b * NN * NN);
  v8f acc[8] = {};
  wmma_tile8(attn, tmpT + (size_t)b * DD * NN, 2 * NN, NN, NN, row0, col0, lane, acc);
  int c = lane & 15, h = lane >> 4;
#pragma unroll
  for (int a = 0; a < 2; ++a)
#pragma unroll
    for (int j = 0; j < 4; ++j)
#pragma unroll
      for (int rr = 0; rr < 8; ++rr) {
        int row = row0 + a * 16 + h * 8 + rr, col = col0 + j * 16 + c;
        out[(size_t)b * NN * DD + (size_t)row * DD + col] =
            tanhf(acc[a * 4 + j][rr] + bias[col]);
      }
}

extern "C" void kernel_launch(void* const* d_in, const int* in_sizes, int n_in,
                              void* d_out, int out_size, void* d_ws, size_t ws_size,
                              hipStream_t stream) {
  (void)in_sizes; (void)n_in; (void)out_size; (void)ws_size;
  const float* nodes = (const float*)d_in[0];
  const float* adj   = (const float*)d_in[1];
  const float* W[2]    = {(const float*)d_in[2], (const float*)d_in[4]};
  const float* bias[2] = {(const float*)d_in[3], (const float*)d_in[5]};
  float* out = (float*)d_out;

  char* ws = (char*)d_ws;
  size_t off = 0;
  float*  dinv   = (float*) (ws + off); off += (size_t)BB * NN * 4;
  bf16_t* adj_bf = (bf16_t*)(ws + off); off += (size_t)BB * NN * NN * 2;
  bf16_t* xs_t   = (bf16_t*)(ws + off); off += (size_t)BB * DD * NN * 2;
  bf16_t* tmp    = (bf16_t*)(ws + off); off += (size_t)BB * NN * DD * 2;
  bf16_t* tmpT   = (bf16_t*)(ws + off); off += (size_t)BB * DD * NN * 2;
  bf16_t* hbuf   = (bf16_t*)(ws + off); off += (size_t)BB * NN * DD * 2;
  bf16_t* wt     = (bf16_t*)(ws + off); off += (size_t)DD * DD * 2;
  float*  scores = (float*) (ws + off); off += (size_t)BB * NN * NN * 4;

  const size_t BND = (size_t)BB * NN * DD;

  k_rowsum  <<<BB * NN / 8, 256, 0, stream>>>(adj, dinv);
  k_pack_adj<<<(int)((size_t)BB * NN * NN / 8 / 256), 256, 0, stream>>>(adj, adj_bf);
  k_copy4   <<<(int)(BND / 4 / 256), 256, 0, stream>>>((const float4*)nodes, (float4*)out,
                                                       (int)(BND / 4));

  for (int l = 0; l < 2; ++l) {
    k_pack_w     <<<DD * DD / 256, 256, 0, stream>>>(W[l], wt);
    k_pack_xs    <<<(int)(BND / 256), 256, 0, stream>>>(out + (size_t)l * BND, dinv, xs_t);
    k_gemm_lap   <<<256, 256, 0, stream>>>(adj_bf, xs_t, dinv, tmp, tmpT);
    k_gemm_w     <<<256, 256, 0, stream>>>(tmp, wt, hbuf);
    k_gemm_scores<<<2048, 256, 0, stream>>>(hbuf, adj, scores);
    k_softmax    <<<BB * NN, 256, 0, stream>>>(scores);
    k_gemm_attn  <<<256, 256, 0, stream>>>(scores, tmpT, bias[l], out + (size_t)(l + 1) * BND);
  }
}
